// ParLinear_61117384622669
// MI455X (gfx1250) — matrix operand order repaired and checked by reading, not compiled, not run on hardware
//
#include <hip/hip_runtime.h>
#include <cstdint>
#include <cstddef>

// ---------------------------------------------------------------------------
// out[b, c] = sum_k x[b,k] * mat[c,k] + bias[c]
// M=4096, N=32000, K=1024, fp32 in/out.
// Pass 1: fp32 -> bf16 copies of x and mat into workspace (one-time convert).
// Pass 2: bf16 WMMA GEMM, double-buffered LDS fed by GLOBAL_LOAD_ASYNC_TO_LDS,
//         fp32 accumulate, bias fused in epilogue.
// Fallback (workspace too small): fused convert-in-loop kernel from round 1.
// ---------------------------------------------------------------------------

#define EMBED   1024
#define CLASSES 32000
#define BATCHSZ 4096

#define BM 128
#define BN 128

typedef __attribute__((ext_vector_type(16))) __bf16 v16bf;
typedef __attribute__((ext_vector_type(8)))  float  v8f;
typedef int vi4 __attribute__((vector_size(16)));   // matches builtin's V4i

union FragU { uint4 u[2]; v16bf v; };

// fp32 -> 2x bf16 packed, round-to-nearest-even (pure integer VALU).
static __device__ __forceinline__ uint32_t pk_bf16(float a, float b) {
    uint32_t ua = __float_as_uint(a);
    uint32_t ub = __float_as_uint(b);
    ua = (ua + 0x7FFFu + ((ua >> 16) & 1u)) >> 16;
    ub = (ub + 0x7FFFu + ((ub >> 16) & 1u)) >> 16;
    return (ua & 0xFFFFu) | (ub << 16);
}

// ---------------- CDNA5 async-copy helpers ----------------
#define AS1 __attribute__((address_space(1)))
#define AS3 __attribute__((address_space(3)))

#if __has_builtin(__builtin_amdgcn_global_load_async_to_lds_b128)
static __device__ __forceinline__ void async_cp_b128(const void* g, void* l) {
    __builtin_amdgcn_global_load_async_to_lds_b128(
        (AS1 vi4*)(g), (AS3 vi4*)(l), 0, 0);
}
#else
static __device__ __forceinline__ void async_cp_b128(const void* g, void* l) {
    unsigned           loff = (unsigned)(size_t)(AS3 void*)(l);
    unsigned long long gadr = (unsigned long long)(size_t)(g);
    asm volatile("global_load_async_to_lds_b128 %0, %1, off"
                 :: "v"(loff), "v"(gadr) : "memory");
}
#endif

#if __has_builtin(__builtin_amdgcn_s_wait_asynccnt)
#define WAIT_ASYNC0() __builtin_amdgcn_s_wait_asynccnt(0)
#else
#define WAIT_ASYNC0() asm volatile("s_wait_asynccnt 0x0" ::: "memory")
#endif

// ===========================================================================
// Pass 1: fp32 -> bf16 convert (vectorized: 1 float4 in, 4 bf16 out / thread)
// ===========================================================================
__global__ __launch_bounds__(256)
void cvt_f32_to_bf16(const float* __restrict__ src,
                     unsigned short* __restrict__ dst, int n4) {
    int i = blockIdx.x * blockDim.x + threadIdx.x;
    if (i < n4) {
        float4 v = ((const float4*)src)[i];
        ((uint2*)dst)[i] = make_uint2(pk_bf16(v.x, v.y), pk_bf16(v.z, v.w));
    }
}

// ===========================================================================
// Pass 2: bf16 WMMA GEMM with async double-buffered LDS staging
//   block tile 128x128, K-stage 64, 8 wave32 waves (2M x 4N),
//   per wave: 4x2 v_wmma_f32_16x16x32_bf16 accumulators.
// ===========================================================================
#define BK2      64
#define LDT2     72     // bf16 elems per LDS row (144 B: conflict-free b128)
#define LDSROWB  144

__global__ __launch_bounds__(256)
void gemm_bf16_wmma_async(const unsigned short* __restrict__ xb,
                          const unsigned short* __restrict__ matb,
                          const float* __restrict__ bias,
                          float* __restrict__ out) {
    __shared__ __align__(16) unsigned short As[2][BM * LDT2];  // 2 x 18 KB
    __shared__ __align__(16) unsigned short Bs[2][BN * LDT2];  // 2 x 18 KB

    const int tid  = threadIdx.x;
    const int lane = tid & 31;
    const int wave = tid >> 5;     // 0..7
    const int wm   = wave & 1;     // 2 waves along M (64 rows each)
    const int wn   = wave >> 1;    // 4 waves along N (32 cols each)
    const int ll   = lane & 15;
    const int lh   = lane >> 4;

    const int m0 = blockIdx.y * BM;
    const int n0 = blockIdx.x * BN;

    v8f acc[4][2];
    const v8f vzero = {0.f, 0.f, 0.f, 0.f, 0.f, 0.f, 0.f, 0.f};
#pragma unroll
    for (int i = 0; i < 4; ++i)
#pragma unroll
        for (int j = 0; j < 2; ++j)
            acc[i][j] = vzero;

    // Staging geometry: tile = 128 rows x 64 bf16 = 128 B/row = 8 x b128/row.
    // 1024 b128 per tile, 256 threads -> 4 async b128 per thread per tile.
    const int r0 = tid >> 3;            // 0..31, rows r0 + {0,32,64,96}
    const int cb = (tid & 7) << 4;      // byte column within row: 0..112

    const char* gx = (const char*)(xb   + (size_t)m0 * EMBED);
    const char* gb = (const char*)(matb + (size_t)n0 * EMBED);

    auto issue = [&](int buf, int kt) {
#pragma unroll
        for (int i = 0; i < 4; ++i) {
            const int    r    = r0 + i * 32;
            const size_t goff = ((size_t)r * EMBED + kt) * 2 + cb;
            const int    loff = r * LDSROWB + cb;
            async_cp_b128(gx + goff, (char*)(&As[buf][0]) + loff);
            async_cp_b128(gb + goff, (char*)(&Bs[buf][0]) + loff);
        }
    };

    issue(0, 0);                        // prologue: stage 0 in flight

    const int NSTAGE = EMBED / BK2;     // 16
    for (int s = 0; s < NSTAGE; ++s) {
        WAIT_ASYNC0();                  // own-buffer async copies complete
        __syncthreads();                // all waves' data visible; prev compute done
        if (s + 1 < NSTAGE)
            issue((s + 1) & 1, (s + 1) * BK2);   // overlap next stage

        const unsigned short* Ab = &As[s & 1][0];
        const unsigned short* Bb = &Bs[s & 1][0];
#pragma unroll
        for (int ks = 0; ks < BK2; ks += 32) {
            // A frag (16x32 MxK, 16-bit): lanes 0-15 K={0..7,16..23},
            // lanes 16-31 K={8..15,24..31}; two b128 LDS reads.
            v16bf a[4];
#pragma unroll
            for (int fm = 0; fm < 4; ++fm) {
                const unsigned short* ap =
                    Ab + (wm * 64 + fm * 16 + ll) * LDT2 + ks + (lh ? 8 : 0);
                FragU fa;
                fa.u[0] = *(const uint4*)(ap);
                fa.u[1] = *(const uint4*)(ap + 16);
                a[fm] = fa.v;
            }
            // B frag (32x16 KxN): lanes 0-15 K={0..7,16..23},
            // lanes 16-31 K={8..15,24..31}; two b128 LDS reads.
            v16bf b[2];
#pragma unroll
            for (int fn = 0; fn < 2; ++fn) {
                const unsigned short* bp =
                    Bb + (wn * 32 + fn * 16 + ll) * LDT2 + ks + (lh ? 8 : 0);
                FragU fb;
                fb.u[0] = *(const uint4*)(bp);
                fb.u[1] = *(const uint4*)(bp + 16);
                b[fn] = fb.v;
            }
#pragma unroll
            for (int fm = 0; fm < 4; ++fm)
#pragma unroll
                for (int fn = 0; fn < 2; ++fn)
                    acc[fm][fn] = __builtin_amdgcn_wmma_f32_16x16x32_bf16(
                        false, a[fm], false, b[fn],
                        (short)0, acc[fm][fn], false, false);
        }
    }

    // Epilogue: bias add + fp32 store.
    // C/D layout: lane l, VGPR r -> M = r + 8*(l>=16), N = l%16.
#pragma unroll
    for (int fm = 0; fm < 4; ++fm) {
#pragma unroll
        for (int fn = 0; fn < 2; ++fn) {
            const int col  = n0 + wn * 32 + fn * 16 + ll;
            const int row0 = m0 + wm * 64 + fm * 16 + lh * 8;
            const float bv = bias[col];
            float* op = out + (size_t)row0 * CLASSES + col;
#pragma unroll
            for (int r = 0; r < 8; ++r)
                op[(size_t)r * CLASSES] = acc[fm][fn][r] + bv;
        }
    }
}

// ===========================================================================
// Fallback: fused fp32->bf16-in-loop WMMA kernel (round 1), used only if the
// workspace is too small for the bf16 copies.
// ===========================================================================
#define BKF 32
#define LDTF 40

__global__ __launch_bounds__(256)
void parlinear_wmma_fused(const float* __restrict__ x,
                          const float* __restrict__ mat,
                          const float* __restrict__ bias,
                          float* __restrict__ out) {
    __shared__ __align__(16) unsigned short As[BM * LDTF];
    __shared__ __align__(16) unsigned short Bs[BN * LDTF];

    const int tid  = threadIdx.x;
    const int lane = tid & 31;
    const int wave = tid >> 5;
    const int wm   = wave & 1;
    const int wn   = wave >> 1;
    const int ll   = lane & 15;
    const int lh   = lane >> 4;

    const int m0 = blockIdx.y * BM;
    const int n0 = blockIdx.x * BN;

    v8f acc[4][2];
    const v8f vzero = {0.f, 0.f, 0.f, 0.f, 0.f, 0.f, 0.f, 0.f};
#pragma unroll
    for (int i = 0; i < 4; ++i)
#pragma unroll
        for (int j = 0; j < 2; ++j)
            acc[i][j] = vzero;

    for (int kt = 0; kt < EMBED; kt += BKF) {
        __syncthreads();
#pragma unroll
        for (int i = 0; i < 4; ++i) {
            const int fid = tid + i * 256;
            const int r   = fid >> 3;
            const int c4  = (fid & 7) << 2;
            const float4 av = *(const float4*)(&x  [(size_t)(m0 + r) * EMBED + kt + c4]);
            const float4 bv = *(const float4*)(&mat[(size_t)(n0 + r) * EMBED + kt + c4]);
            *(uint2*)(&As[r * LDTF + c4]) = make_uint2(pk_bf16(av.x, av.y), pk_bf16(av.z, av.w));
            *(uint2*)(&Bs[r * LDTF + c4]) = make_uint2(pk_bf16(bv.x, bv.y), pk_bf16(bv.z, bv.w));
        }
        __syncthreads();

        v16bf a[4];
#pragma unroll
        for (int fm = 0; fm < 4; ++fm) {
            const unsigned short* ap = &As[(wm * 64 + fm * 16 + ll) * LDTF + (lh ? 8 : 0)];
            FragU fa;
            fa.u[0] = *(const uint4*)(ap);
            fa.u[1] = *(const uint4*)(ap + 16);
            a[fm] = fa.v;
        }
        v16bf b[2];
#pragma unroll
        for (int fn = 0; fn < 2; ++fn) {
            const unsigned short* bp = &Bs[(wn * 32 + fn * 16 + ll) * LDTF + (lh ? 8 : 0)];
            FragU fb;
            fb.u[0] = *(const uint4*)(bp);
            fb.u[1] = *(const uint4*)(bp + 16);
            b[fn] = fb.v;
        }
#pragma unroll
        for (int fm = 0; fm < 4; ++fm)
#pragma unroll
            for (int fn = 0; fn < 2; ++fn)
                acc[fm][fn] = __builtin_amdgcn_wmma_f32_16x16x32_bf16(
                    false, a[fm], false, b[fn], (short)0, acc[fm][fn], false, false);
    }

#pragma unroll
    for (int fm = 0; fm < 4; ++fm) {
#pragma unroll
        for (int fn = 0; fn < 2; ++fn) {
            const int col  = n0 + wn * 32 + fn * 16 + ll;
            const int row0 = m0 + wm * 64 + fm * 16 + lh * 8;
            const float bv = bias[col];
            float* op = out + (size_t)row0 * CLASSES + col;
#pragma unroll
            for (int r = 0; r < 8; ++r)
                op[(size_t)r * CLASSES] = acc[fm][fn][r] + bv;
        }
    }
}

// ===========================================================================
extern "C" void kernel_launch(void* const* d_in, const int* in_sizes, int n_in,
                              void* d_out, int out_size, void* d_ws, size_t ws_size,
                              hipStream_t stream) {
    const float* x    = (const float*)d_in[0];   // (4096, 1, 1024)
    const float* mat  = (const float*)d_in[1];   // (1, 32000, 1024)
    const float* bias = (const float*)d_in[2];   // (1, 32000)
    float*       out  = (float*)d_out;           // (4096, 32000)

    const size_t xElems   = (size_t)BATCHSZ * EMBED;     // 4,194,304
    const size_t matElems = (size_t)CLASSES * EMBED;     // 32,768,000
    const size_t needWs   = (xElems + matElems) * sizeof(unsigned short);

    dim3 grid(CLASSES / BN, BATCHSZ / BM);               // (250, 32)
    dim3 block(256);

    if (ws_size >= needWs) {
        unsigned short* xb   = (unsigned short*)d_ws;
        unsigned short* matb = xb + xElems;

        const int xN4   = (int)(xElems / 4);
        const int matN4 = (int)(matElems / 4);
        cvt_f32_to_bf16<<<(xN4 + 255) / 256, 256, 0, stream>>>(x, xb, xN4);
        cvt_f32_to_bf16<<<(matN4 + 255) / 256, 256, 0, stream>>>(mat, matb, matN4);

        gemm_bf16_wmma_async<<<grid, block, 0, stream>>>(xb, matb, bias, out);
    } else {
        parlinear_wmma_fused<<<grid, block, 0, stream>>>(x, mat, bias, out);
    }
}
